// TransLayer_27633819583240
// MI455X (gfx1250) — matrix operand order repaired and checked by reading, not compiled, not run on hardware
//
#include <hip/hip_runtime.h>
#include <hip/hip_bf16.h>

// ---------------- constants (match reference) ----------------
#define DIMC      256
#define HEADS     8
#define DIM_HEAD  32
#define LMK       128          // LANDMARKS
#define SEQ       4000
#define NPAD      4096
#define PADR      96           // -4000 % 128
#define KW        33           // conv kernel
#define INNER     256          // HEADS*DIM_HEAD
#define QSCALE    0.1767766952966369f   // 32^-0.5
#define LN_EPS    1e-5f

typedef unsigned short u16;
typedef __attribute__((ext_vector_type(16))) __bf16 v16bf;
typedef __attribute__((ext_vector_type(8)))  float  v8f;

// ---------------- helpers ----------------
__device__ __forceinline__ u16 f2bf(float f) {
  unsigned int u = __float_as_uint(f);
  u += 0x7fffu + ((u >> 16) & 1u);          // round-to-nearest-even
  return (u16)(u >> 16);
}

// =====================================================================
// GEMM kernel 1: B stored N-major (element (k,n) at B + n*sbn + k).
// Both fragments are 2 x b128 loads per lane. Used for the QK^T logits.
// One wave per 16x16 tile, 8 waves/block, partial last block allowed.
// =====================================================================
__global__ void k_gemm_bn(const u16* __restrict__ A, long ba, int sam,
                          const u16* __restrict__ B, long bb, int sbn,
                          float* __restrict__ C, long bc,
                          int M, int N, int K) {
  const int lane = threadIdx.x & 31;
  const int wave = threadIdx.x >> 5;
  const int l = lane & 15, hi = (lane >> 4) & 1;
  const int tilesN = N >> 4;
  const long totalTiles = (long)(M >> 4) * tilesN;
  long tile = (long)blockIdx.x * 8 + wave;
  if (tile >= totalTiles) return;               // uniform per-wave exit
  const int tm = (int)(tile / tilesN);
  const int tn = (int)(tile % tilesN);
  const int batch = blockIdx.y;

  const u16* Ap = A + (long)batch * ba + (long)(tm * 16 + l) * sam;
  const u16* Bp = B + (long)batch * bb + (long)(tn * 16 + l) * sbn;

  v8f acc = {};
  for (int kk = 0; kk < K; kk += 32) {
    union { uint4 q[2]; v16bf v; } av, bv;
    av.q[0] = *(const uint4*)(Ap + kk + hi * 16);       // K = hi*16..+7
    av.q[1] = *(const uint4*)(Ap + kk + hi * 16 + 8);   // K = hi*16+8..
    bv.q[0] = *(const uint4*)(Bp + kk + hi * 16);       // K = hi*16..+7
    bv.q[1] = *(const uint4*)(Bp + kk + hi * 16 + 8);   // K = hi*16+8..
    acc = __builtin_amdgcn_wmma_f32_16x16x32_bf16(false, av.v, false, bv.v,
                                                  (short)0, acc, false, false);
  }
  float* Cb = C + (long)batch * bc;
  const int col = tn * 16 + l;
  #pragma unroll
  for (int r = 0; r < 8; ++r) {
    int row = tm * 16 + hi * 8 + r;
    Cb[(long)row * N + col] = acc[r];
  }
}

// =====================================================================
// GEMM kernel 2: B stored row-major K x N (element (k,n) at B + k*sbk + n).
// Block = 8 waves x 8 M-tiles of one N-group of NT tiles (16*NT columns).
// The 32 x (16*NT) B tile is cooperatively staged TRANSPOSED through LDS
// (coalesced b32 global reads along N, ds stores, then per-lane b128
// fragment reads in A's contiguous k = 16*hi + s order), shared by all 8 waves. Each wave
// holds NT accumulators and reuses one A fragment for NT WMMAs per K-step.
// Requires M % 128 == 0 and N % (16*NT) == 0 (true at every call site).
// EPI 0: C = scale*acc (+ optional bf16 mirror)
// EPI 1: final projection: rows>=padr -> dfin = xres + acc + bias
// =====================================================================
template <int EPI, int NT>
__global__ void k_gemm_tb(const u16* __restrict__ A, long ba, int sam,
                          const u16* __restrict__ B, long bb, int sbk,
                          float* __restrict__ C, long bc, u16* __restrict__ Cbf,
                          int N, int K, float scale,
                          const float* __restrict__ xres, const float* __restrict__ bias,
                          float* __restrict__ dfin, int padr) {
  __shared__ alignas(16) u16 ldsB[16 * NT][40];   // [n][k], 80B rows (16B aligned)
  const int tid = threadIdx.x;
  const int lane = tid & 31;
  const int wave = tid >> 5;
  const int l = lane & 15, hi = (lane >> 4) & 1;
  const int groups = N / (16 * NT);
  const int gn = blockIdx.x % groups;
  const int tm = (blockIdx.x / groups) * 8 + wave;
  const int batch = blockIdx.y;

  const u16* Ap = A + (long)batch * ba + (long)(tm * 16 + l) * sam;
  const u16* Bt = B + (long)batch * bb + gn * (16 * NT);

  // staging map: pair p = tid + j*256 (j<NT) -> k = p/(8*NT), n = 2*(p%(8*NT))
  const int sk0 = tid / (8 * NT);
  const int sn0 = (tid % (8 * NT)) * 2;

  v8f zero = {};
  v8f acc[NT];
  #pragma unroll
  for (int i = 0; i < NT; ++i) acc[i] = zero;

  for (int kk = 0; kk < K; kk += 32) {
    // ---- stage B tile transposed into LDS ----
    #pragma unroll
    for (int j = 0; j < NT; ++j) {
      int p = tid + j * 256;
      int k = p / (8 * NT);
      int sn = (p % (8 * NT)) * 2;
      unsigned int w = *(const unsigned int*)(Bt + (long)(kk + k) * sbk + sn);
      ldsB[sn][k]     = (u16)(w & 0xffffu);
      ldsB[sn + 1][k] = (u16)(w >> 16);
    }
    __syncthreads();
    // ---- prefetch next K-tile while WMMA runs ----
    if (kk + 32 < K) {
      __builtin_prefetch(Bt + (long)(kk + 32 + sk0) * sbk + sn0, 0, 1);
      __builtin_prefetch(Ap + kk + 32, 0, 1);
    }
    // ---- one A fragment, NT B fragments, NT WMMAs ----
    union { uint4 q[2]; v16bf v; } av;
    av.q[0] = *(const uint4*)(Ap + kk + hi * 16);
    av.q[1] = *(const uint4*)(Ap + kk + hi * 16 + 8);
    #pragma unroll
    for (int nt = 0; nt < NT; ++nt) {
      union { uint4 q[2]; v16bf v; } bv;
      const u16* Lp = &ldsB[nt * 16 + l][hi * 16];   // 16 contiguous K values
      bv.q[0] = ((const uint4*)Lp)[0];
      bv.q[1] = ((const uint4*)Lp)[1];
      acc[nt] = __builtin_amdgcn_wmma_f32_16x16x32_bf16(false, av.v, false, bv.v,
                                                        (short)0, acc[nt], false, false);
    }
    __syncthreads();
  }

  #pragma unroll
  for (int nt = 0; nt < NT; ++nt) {
    const int col = (gn * NT + nt) * 16 + l;
    if (EPI == 0) {
      float* Cb = C + (long)batch * bc;
      #pragma unroll
      for (int r = 0; r < 8; ++r) {
        int row = tm * 16 + hi * 8 + r;
        float v = acc[nt][r] * scale;
        long idx = (long)row * N + col;
        Cb[idx] = v;
        if (Cbf) Cbf[(long)batch * bc + idx] = f2bf(v);
      }
    } else {
      #pragma unroll
      for (int r = 0; r < 8; ++r) {
        int row = tm * 16 + hi * 8 + r;
        if (row >= padr) {
          long o = (long)(row - padr) * N + col;
          dfin[o] = xres[o] + acc[nt][r] * scale + bias[col];
        }
      }
    }
  }
}

// ---------------- LayerNorm + front-pad ----------------
__global__ void k_layernorm(const float* __restrict__ x, const float* __restrict__ g,
                            const float* __restrict__ beta,
                            float* __restrict__ xpf, u16* __restrict__ xpb) {
  int row = blockIdx.x;
  int t = threadIdx.x;
  long o = (long)row * DIMC + t;
  if (row < PADR) { xpf[o] = 0.f; xpb[o] = f2bf(0.f); return; }
  float v = x[(long)(row - PADR) * DIMC + t];
  __shared__ float s1[256], s2[256];
  s1[t] = v; s2[t] = v * v;
  __syncthreads();
  for (int s = 128; s > 0; s >>= 1) {
    if (t < s) { s1[t] += s1[t + s]; s2[t] += s2[t + s]; }
    __syncthreads();
  }
  float mu  = s1[0] * (1.0f / DIMC);
  float var = s2[0] * (1.0f / DIMC) - mu * mu;
  float y = (v - mu) * rsqrtf(var + LN_EPS) * g[t] + beta[t];
  xpf[o] = y; xpb[o] = f2bf(y);
}

// ---------------- f32 -> bf16 convert ----------------
__global__ void k_cvt_bf16(const float* __restrict__ in, u16* __restrict__ out, long n) {
  long i = (long)blockIdx.x * blockDim.x + threadIdx.x;
  if (i < n) out[i] = f2bf(in[i]);
}

// ---------------- split qkv into per-head q(scaled)/k/v ----------------
__global__ void k_split_heads(const float* __restrict__ qkv,
                              float* __restrict__ qf, u16* __restrict__ qb,
                              float* __restrict__ kf, u16* __restrict__ kb,
                              float* __restrict__ vf, u16* __restrict__ vb) {
  long i = (long)blockIdx.x * 256 + threadIdx.x;
  if (i >= (long)NPAD * INNER) return;
  int n = (int)(i >> 8), c = (int)(i & 255);
  int h = c >> 5, d = c & 31;
  long ho = ((long)h * NPAD + n) * DIM_HEAD + d;
  long ro = (long)n * 768 + c;
  float q = qkv[ro] * QSCALE;
  float k = qkv[ro + 256];
  float v = qkv[ro + 512];
  qf[ho] = q; qb[ho] = f2bf(q);
  kf[ho] = k; kb[ho] = f2bf(k);
  vf[ho] = v; vb[ho] = f2bf(v);
}

// ---------------- landmark means (32-token groups) ----------------
__global__ void k_landmark(const float* __restrict__ src, float* __restrict__ df,
                           u16* __restrict__ db) {
  int j = blockIdx.x, h = blockIdx.y, d = threadIdx.x;
  const float* p = src + ((long)h * NPAD + (long)j * 32) * DIM_HEAD + d;
  float s = 0.f;
  #pragma unroll
  for (int t = 0; t < 32; ++t) s += p[t * DIM_HEAD];
  s *= (1.0f / 32.0f);
  long o = ((long)h * LMK + j) * DIM_HEAD + d;
  df[o] = s; db[o] = f2bf(s);
}

// ---------------- row softmax (in-place f32, emits bf16 mirror) ----------------
__global__ void k_softmax(float* __restrict__ d, u16* __restrict__ obf, int L) {
  long base = (long)blockIdx.x * L;
  int t = threadIdx.x;
  __shared__ float red[256];
  float m = -3.4e38f;
  for (int i = t; i < L; i += 256) m = fmaxf(m, d[base + i]);
  red[t] = m; __syncthreads();
  for (int s = 128; s > 0; s >>= 1) { if (t < s) red[t] = fmaxf(red[t], red[t + s]); __syncthreads(); }
  m = red[0]; __syncthreads();
  float sum = 0.f;
  for (int i = t; i < L; i += 256) sum += __expf(d[base + i] - m);
  red[t] = sum; __syncthreads();
  for (int s = 128; s > 0; s >>= 1) { if (t < s) red[t] += red[t + s]; __syncthreads(); }
  float inv = 1.0f / red[0];
  for (int i = t; i < L; i += 256) {
    float e = __expf(d[base + i] - m) * inv;
    d[base + i] = e; obf[base + i] = f2bf(e);
  }
}

// ---------------- pinv scalars ----------------
__global__ void k_zero_u32(unsigned int* p, int n) {
  int i = threadIdx.x; if (i < n) p[i] = 0u;
}
__global__ void k_pinv_norms(const float* __restrict__ a2, unsigned int* __restrict__ scal) {
  int tid = blockIdx.x * 256 + threadIdx.x;
  if (tid >= HEADS * LMK) return;
  int h = tid >> 7, i = tid & 127;
  const float* base = a2 + ((long)h << 14);
  float rs = 0.f, cs = 0.f;
  for (int j = 0; j < LMK; ++j) {
    rs += fabsf(base[i * LMK + j]);
    cs += fabsf(base[j * LMK + i]);
  }
  atomicMax(&scal[0], __float_as_uint(rs));   // col = max row-sum
  atomicMax(&scal[1], __float_as_uint(cs));   // row = max col-sum
}
__global__ void k_zinit(const float* __restrict__ a2, const unsigned int* __restrict__ scal,
                        float* __restrict__ zf, u16* __restrict__ zb) {
  long i = (long)blockIdx.x * 256 + threadIdx.x;
  if (i >= (long)HEADS * LMK * LMK) return;
  int h = (int)(i >> 14), r = (int)((i >> 7) & 127), c = (int)(i & 127);
  float inv = 1.0f / (__uint_as_float(scal[0]) * __uint_as_float(scal[1]));
  float v = a2[((long)h << 14) + ((long)c << 7) + r] * inv;   // transpose
  zf[i] = v; zb[i] = f2bf(v);
}
__global__ void k_alpha_i_minus(const float* __restrict__ in, float* __restrict__ of,
                                u16* __restrict__ ob, float alpha) {
  long i = (long)blockIdx.x * 256 + threadIdx.x;
  if (i >= (long)HEADS * LMK * LMK) return;
  int r = (int)((i >> 7) & 127), c = (int)(i & 127);
  float v = ((r == c) ? alpha : 0.f) - in[i];
  of[i] = v; ob[i] = f2bf(v);
}

// ---------------- depthwise conv-33 residual + head->flat repack ----------------
__global__ void k_conv_res(const float* __restrict__ outh, const float* __restrict__ vf,
                           const float* __restrict__ kern, u16* __restrict__ flatb) {
  long i = (long)blockIdx.x * 256 + threadIdx.x;
  if (i >= (long)HEADS * NPAD * DIM_HEAD) return;
  int h = (int)(i / (NPAD * DIM_HEAD));
  int rem = (int)(i % (NPAD * DIM_HEAD));
  int n = rem / DIM_HEAD, d = rem % DIM_HEAD;
  float acc = outh[i];
  const float* vp = vf + (long)h * NPAD * DIM_HEAD + d;
  const float* kp = kern + h * KW;
  #pragma unroll
  for (int t = 0; t < KW; ++t) {
    int m = n + t - KW / 2;
    if (m >= 0 && m < NPAD) acc += vp[(long)m * DIM_HEAD] * kp[t];
  }
  flatb[(long)n * INNER + h * DIM_HEAD + d] = f2bf(acc);
}

__global__ void k_tail(float* __restrict__ d) {
  if (threadIdx.x == 0) { d[0] = (float)PADR; d[1] = (float)SEQ; }
}

// ---------------- host side ----------------
static inline void* wsalloc(char*& cur, size_t bytes) {
  void* p = (void*)cur;
  cur += (bytes + 255) & ~(size_t)255;
  return p;
}
// B row-major KxN (sbn==1) path; M % 128 == 0, N % (16*NT) == 0
template <int NT>
static inline void gemm_tb(hipStream_t s, const u16* A, long ba, int sam,
                           const u16* B, long bb, int sbk,
                           float* C, long bc, u16* Cbf,
                           int M, int N, int K, int batch, float scale) {
  int blocks = (M / 128) * (N / (16 * NT));
  k_gemm_tb<0, NT><<<dim3(blocks, batch), 256, 0, s>>>(A, ba, sam, B, bb, sbk, C, bc, Cbf,
                                                       N, K, scale, nullptr, nullptr, nullptr, 0);
}
// B N-major (sbk==1) path
static inline void gemm_bn(hipStream_t s, const u16* A, long ba, int sam,
                           const u16* B, long bb, int sbn,
                           float* C, long bc, int M, int N, int K, int batch) {
  long tiles = (long)(M / 16) * (N / 16);
  int blocks = (int)((tiles + 7) / 8);
  k_gemm_bn<<<dim3(blocks, batch), 256, 0, s>>>(A, ba, sam, B, bb, sbn, C, bc, M, N, K);
}

extern "C" void kernel_launch(void* const* d_in, const int* in_sizes, int n_in,
                              void* d_out, int out_size, void* d_ws, size_t ws_size,
                              hipStream_t stream) {
  (void)in_sizes; (void)n_in; (void)out_size; (void)ws_size;
  const float* x     = (const float*)d_in[0];
  const float* ln_g  = (const float*)d_in[1];
  const float* ln_b  = (const float*)d_in[2];
  const float* w_qkv = (const float*)d_in[3];
  const float* w_out = (const float*)d_in[4];
  const float* b_out = (const float*)d_in[5];
  const float* res_k = (const float*)d_in[6];
  float* out = (float*)d_out;

  const long SQ = (long)LMK * LMK;
  char* cur = (char*)d_ws;
  float* xpf  = (float*)wsalloc(cur, (size_t)NPAD * DIMC * 4);
  u16*   xpb  = (u16*)  wsalloc(cur, (size_t)NPAD * DIMC * 2);
  u16*   wqb  = (u16*)  wsalloc(cur, (size_t)DIMC * 768 * 2);
  u16*   wob  = (u16*)  wsalloc(cur, (size_t)DIMC * DIMC * 2);
  float* qkvf = (float*)wsalloc(cur, (size_t)NPAD * 768 * 4);
  float* qf   = (float*)wsalloc(cur, (size_t)NPAD * INNER * 4);
  float* kf   = (float*)wsalloc(cur, (size_t)NPAD * INNER * 4);
  float* vf   = (float*)wsalloc(cur, (size_t)NPAD * INNER * 4);
  u16*   qb   = (u16*)  wsalloc(cur, (size_t)NPAD * INNER * 2);
  u16*   kb   = (u16*)  wsalloc(cur, (size_t)NPAD * INNER * 2);
  u16*   vb   = (u16*)  wsalloc(cur, (size_t)NPAD * INNER * 2);
  float* qlf  = (float*)wsalloc(cur, (size_t)HEADS * LMK * DIM_HEAD * 4);
  float* klf  = (float*)wsalloc(cur, (size_t)HEADS * LMK * DIM_HEAD * 4);
  u16*   qlb  = (u16*)  wsalloc(cur, (size_t)HEADS * LMK * DIM_HEAD * 2);
  u16*   klb  = (u16*)  wsalloc(cur, (size_t)HEADS * LMK * DIM_HEAD * 2);
  float* at1f = (float*)wsalloc(cur, (size_t)HEADS * NPAD * LMK * 4);
  u16*   at1b = (u16*)  wsalloc(cur, (size_t)HEADS * NPAD * LMK * 2);
  float* at2f = (float*)wsalloc(cur, (size_t)HEADS * SQ * 4);
  u16*   at2b = (u16*)  wsalloc(cur, (size_t)HEADS * SQ * 2);
  float* at3f = (float*)wsalloc(cur, (size_t)HEADS * LMK * NPAD * 4);
  u16*   at3b = (u16*)  wsalloc(cur, (size_t)HEADS * LMK * NPAD * 2);
  float* z0f  = (float*)wsalloc(cur, (size_t)HEADS * SQ * 4);
  u16*   z0b  = (u16*)  wsalloc(cur, (size_t)HEADS * SQ * 2);
  float* z1f  = (float*)wsalloc(cur, (size_t)HEADS * SQ * 4);
  u16*   z1b  = (u16*)  wsalloc(cur, (size_t)HEADS * SQ * 2);
  float* azf  = (float*)wsalloc(cur, (size_t)HEADS * SQ * 4);
  u16*   azb  = (u16*)  wsalloc(cur, (size_t)HEADS * SQ * 2);
  float* t1f  = (float*)wsalloc(cur, (size_t)HEADS * SQ * 4);
  u16*   t1b  = (u16*)  wsalloc(cur, (size_t)HEADS * SQ * 2);
  float* t2f  = (float*)wsalloc(cur, (size_t)HEADS * SQ * 4);
  u16*   t2b  = (u16*)  wsalloc(cur, (size_t)HEADS * SQ * 2);
  float* a12f = (float*)wsalloc(cur, (size_t)HEADS * NPAD * LMK * 4);
  u16*   a12b = (u16*)  wsalloc(cur, (size_t)HEADS * NPAD * LMK * 2);
  float* a3vf = (float*)wsalloc(cur, (size_t)HEADS * LMK * DIM_HEAD * 4);
  u16*   a3vb = (u16*)  wsalloc(cur, (size_t)HEADS * LMK * DIM_HEAD * 2);
  float* othf = (float*)wsalloc(cur, (size_t)HEADS * NPAD * DIM_HEAD * 4);
  u16*   flb  = (u16*)  wsalloc(cur, (size_t)NPAD * INNER * 2);
  unsigned int* scal = (unsigned int*)wsalloc(cur, 64);

  // 1) LayerNorm + pad
  k_layernorm<<<NPAD, 256, 0, stream>>>(x, ln_g, ln_b, xpf, xpb);
  // 2) weight converts
  k_cvt_bf16<<<(DIMC * 768 + 255) / 256, 256, 0, stream>>>(w_qkv, wqb, (long)DIMC * 768);
  k_cvt_bf16<<<(DIMC * DIMC + 255) / 256, 256, 0, stream>>>(w_out, wob, (long)DIMC * DIMC);
  // 3) qkv = xp @ w_qkv   (M=4096, N=768, K=256)
  gemm_tb<4>(stream, xpb, 0, DIMC, wqb, 0, 768, qkvf, 0, nullptr,
             NPAD, 768, DIMC, 1, 1.0f);
  // 4) split into heads (q scaled)
  k_split_heads<<<(NPAD * INNER + 255) / 256, 256, 0, stream>>>(qkvf, qf, qb, kf, kb, vf, vb);
  // 5) landmarks
  k_landmark<<<dim3(LMK, HEADS), DIM_HEAD, 0, stream>>>(qf, qlf, qlb);
  k_landmark<<<dim3(LMK, HEADS), DIM_HEAD, 0, stream>>>(kf, klf, klb);
  // 6) attn1 = softmax(q @ k_l^T)    M=4096,N=128,K=32 ; B = k_l stored N-major
  gemm_bn(stream, qb, (long)NPAD * DIM_HEAD, DIM_HEAD,
          klb, (long)LMK * DIM_HEAD, DIM_HEAD,
          at1f, (long)NPAD * LMK, NPAD, LMK, DIM_HEAD, HEADS);
  k_softmax<<<HEADS * NPAD, 256, 0, stream>>>(at1f, at1b, LMK);
  // 7) attn2 = softmax(q_l @ k_l^T)  M=N=128,K=32
  gemm_bn(stream, qlb, (long)LMK * DIM_HEAD, DIM_HEAD,
          klb, (long)LMK * DIM_HEAD, DIM_HEAD,
          at2f, SQ, LMK, LMK, DIM_HEAD, HEADS);
  k_softmax<<<HEADS * LMK, 256, 0, stream>>>(at2f, at2b, LMK);
  // 8) attn3 = softmax(q_l @ k^T)    M=128,N=4096,K=32 ; B = k stored N-major
  gemm_bn(stream, qlb, (long)LMK * DIM_HEAD, DIM_HEAD,
          kb, (long)NPAD * DIM_HEAD, DIM_HEAD,
          at3f, (long)LMK * NPAD, LMK, NPAD, DIM_HEAD, HEADS);
  k_softmax<<<HEADS * LMK, 256, 0, stream>>>(at3f, at3b, NPAD);
  // 9) pinv init
  k_zero_u32<<<1, 32, 0, stream>>>(scal, 2);
  k_pinv_norms<<<(HEADS * LMK + 255) / 256, 256, 0, stream>>>(at2f, scal);
  k_zinit<<<(int)((HEADS * SQ + 255) / 256), 256, 0, stream>>>(at2f, scal, z0f, z0b);
  // 10) 6 Newton-Schulz iterations: z = 0.25*z@(13I - az@(15I - az@(7I - az)))
  int eb = (int)((HEADS * SQ + 255) / 256);
  for (int it = 0; it < 6; ++it) {
    u16*   zcb = (it & 1) ? z1b : z0b;
    float* znf = (it & 1) ? z0f : z1f;
    u16*   znb = (it & 1) ? z0b : z1b;
    gemm_tb<4>(stream, at2b, SQ, LMK, zcb, SQ, LMK, azf, SQ, azb, LMK, LMK, LMK, HEADS, 1.0f);
    k_alpha_i_minus<<<eb, 256, 0, stream>>>(azf, t1f, t1b, 7.0f);
    gemm_tb<4>(stream, azb, SQ, LMK, t1b, SQ, LMK, t2f, SQ, t2b, LMK, LMK, LMK, HEADS, 1.0f);
    k_alpha_i_minus<<<eb, 256, 0, stream>>>(t2f, t1f, t1b, 15.0f);
    gemm_tb<4>(stream, azb, SQ, LMK, t1b, SQ, LMK, t2f, SQ, t2b, LMK, LMK, LMK, HEADS, 1.0f);
    k_alpha_i_minus<<<eb, 256, 0, stream>>>(t2f, t1f, t1b, 13.0f);
    gemm_tb<4>(stream, zcb, SQ, LMK, t1b, SQ, LMK, znf, SQ, znb, LMK, LMK, LMK, HEADS, 0.25f);
  }
  // final z (even #iters) lives in z0
  // 11) a12 = attn1 @ pinv           M=4096,N=128,K=128
  gemm_tb<4>(stream, at1b, (long)NPAD * LMK, LMK, z0b, SQ, LMK,
             a12f, (long)NPAD * LMK, a12b, NPAD, LMK, LMK, HEADS, 1.0f);
  // 12) a3v = attn3 @ v              M=128,N=32,K=4096 ; B = v row-major K x 32
  gemm_tb<2>(stream, at3b, (long)LMK * NPAD, NPAD, vb, (long)NPAD * DIM_HEAD, DIM_HEAD,
             a3vf, (long)LMK * DIM_HEAD, a3vb, LMK, DIM_HEAD, NPAD, HEADS, 1.0f);
  // 13) out_heads = a12 @ a3v        M=4096,N=32,K=128
  gemm_tb<2>(stream, a12b, (long)NPAD * LMK, LMK, a3vb, (long)LMK * DIM_HEAD, DIM_HEAD,
             othf, (long)NPAD * DIM_HEAD, nullptr, NPAD, DIM_HEAD, LMK, HEADS, 1.0f);
  // 14) att = a12 @ attn3 -> d_out[1024000...]   M=4096,N=4096,K=128 (x8 heads)
  gemm_tb<4>(stream, a12b, (long)NPAD * LMK, LMK, at3b, (long)LMK * NPAD, NPAD,
             out + (long)SEQ * DIMC, (long)NPAD * NPAD, nullptr, NPAD, NPAD, LMK, HEADS, 1.0f);
  // 15) depthwise conv residual, repack heads -> flat bf16
  k_conv_res<<<(HEADS * NPAD * DIM_HEAD + 255) / 256, 256, 0, stream>>>(othf, vf, res_k, flb);
  // 16) final: d_out[0..] = x + flat @ w_out + b_out (rows >= pad)
  k_gemm_tb<1, 4><<<dim3((NPAD / 128) * (DIMC / 64), 1), 256, 0, stream>>>(
      flb, 0, INNER, wob, 0, DIMC, qkvf /*unused*/, 0, nullptr,
      DIMC, INNER, 1.0f, x, b_out, out, PADR);
  // 17) scalar tail outputs (pad, n)
  k_tail<<<1, 32, 0, stream>>>(out + (long)SEQ * DIMC + (long)HEADS * NPAD * NPAD);
}
